// SAGPoolNet_88570815578139
// MI455X (gfx1250) — matrix operand order repaired and checked by reading, not compiled, not run on hardware
//
#include <hip/hip_runtime.h>
#include <math.h>

// ---------------- SAGPoolNet on gfx1250 (MI455X) ----------------
// Dense matmuls -> v_wmma_f32_16x16x32_f16 (f32 accumulate) with
// pre-padded f16 activations and weights repacked into the native
// B-operand layout, so the inner loop is pure b128 loads + wmma.
// Edge aggregation -> one wave32 per edge, float4 gather + f32 atomics.

typedef __attribute__((ext_vector_type(16))) _Float16 v16h;
typedef __attribute__((ext_vector_type(8)))  _Float16 v8h;
typedef __attribute__((ext_vector_type(8)))  float    v8f;

#define NEG_BIG (-3.402823466e38f)

// ---------------- utility kernels ----------------
__global__ void k_fill_f32(float* p, float v, int n) {
  int i = blockIdx.x * blockDim.x + threadIdx.x;
  if (i < n) p[i] = v;
}
__global__ void k_fill_i32(int* p, int v, int n) {
  int i = blockIdx.x * blockDim.x + threadIdx.x;
  if (i < n) p[i] = v;
}

__global__ void k_edge_init(const int* __restrict__ ei, int* __restrict__ src,
                            int* __restrict__ dst, int* __restrict__ em, int E) {
  int e = blockIdx.x * blockDim.x + threadIdx.x;
  if (e < E) { src[e] = ei[e]; dst[e] = ei[E + e]; em[e] = 1; }
}

__global__ void k_deg_accum(const int* __restrict__ dst, const int* __restrict__ em,
                            float* __restrict__ deg, int E) {
  int e = blockIdx.x * blockDim.x + threadIdx.x;
  if (e < E && em[e]) atomicAdd(&deg[dst[e]], 1.0f);
}

__global__ void k_rsqrt(const float* __restrict__ deg, float* __restrict__ dis, int n) {
  int i = blockIdx.x * blockDim.x + threadIdx.x;
  if (i < n) dis[i] = rsqrtf(deg[i]);
}

// activations f32[M,K] -> f16[M,Kp], K zero-padded to Kp (multiple of 32)
__global__ void k_cvt_f16(const float* __restrict__ in, _Float16* __restrict__ out,
                          int K, int Kp, int total /* M*Kp */) {
  int t = blockIdx.x * blockDim.x + threadIdx.x;
  if (t >= total) return;
  int row = t / Kp, k = t - row * Kp;
  out[t] = (k < K) ? (_Float16)in[(size_t)row * K + k] : (_Float16)0.f;
}

// weights f32[K,N] -> f16 packed B-operand layout:
//   out[((tn*Ks + ks)*32 + lane)*16 + j] = W[ks*32 + 8*(lane>>4) + (j&7) + 16*(j>>3)][tn*16 + (lane&15)]
// zero-padded for k >= K or col >= N.
__global__ void k_pack_w(const float* __restrict__ W, _Float16* __restrict__ out,
                         int K, int N, int Ks /* Kp/32 */, int total) {
  int t = blockIdx.x * blockDim.x + threadIdx.x;
  if (t >= total) return;
  int j = t & 15;
  int lane = (t >> 4) & 31;
  int idx = t >> 9;              // (tn*Ks + ks)
  int ks = idx % Ks, tn = idx / Ks;
  int k = ks * 32 + ((lane >> 4) << 3) + (j & 7) + ((j >> 3) << 4);
  int col = tn * 16 + (lane & 15);
  out[t] = (k < K && col < N) ? (_Float16)W[(size_t)k * N + col] : (_Float16)0.f;
}

__global__ void k_pad_bias(const float* __restrict__ in, float* __restrict__ out,
                           int N, int Np) {
  int i = blockIdx.x * blockDim.x + threadIdx.x;
  if (i < Np) out[i] = (i < N) ? in[i] : 0.f;
}

// ---------------- WMMA GEMM: C[M,Np] = A16[M,Kp] @ Wpack (+bias, relu) ----------------
// One wave32 per 16x16 tile; no guards, EXEC stays all-1s, pure b128 loads + wmma.
__global__ void k_gemm_wmma(const _Float16* __restrict__ A, const _Float16* __restrict__ Wp,
                            const float* __restrict__ bias, float* __restrict__ C,
                            int Kp, int Np, int relu) {
  int lane = threadIdx.x & 31;
  int tn = blockIdx.x, tm = blockIdx.y;
  int mlo = lane & 15;      // M index (A) / N index (store)
  int kh  = lane >> 4;      // K-half selector
  int Ks = Kp >> 5;
  const _Float16* Ar = A + (size_t)(tm * 16 + mlo) * Kp + 8 * kh;
  const _Float16* Bp = Wp + ((size_t)tn * Ks * 32 + lane) * 16;
  v8f c = {0.f, 0.f, 0.f, 0.f, 0.f, 0.f, 0.f, 0.f};
  for (int ks = 0; ks < Ks; ++ks) {
    v8h lo = *(const v8h*)(Ar + ks * 32);        // K = ks*32 + 8*kh + [0,8)
    v8h hi = *(const v8h*)(Ar + ks * 32 + 16);   // K = ks*32 + 16 + 8*kh + [0,8)
    v16h b = *(const v16h*)(Bp + (size_t)ks * 512);
    v16h a;
#pragma unroll
    for (int j = 0; j < 8; ++j) { a[j] = lo[j]; a[8 + j] = hi[j]; }
    c = __builtin_amdgcn_wmma_f32_16x16x32_f16(false, a, false, b, (short)0, c, false, false);
  }
  int col = tn * 16 + mlo;
  float bv = bias ? bias[col] : 0.f;
#pragma unroll
  for (int r = 0; r < 8; ++r) {
    float v = c[r] + bv;
    if (relu) v = fmaxf(v, 0.f);
    C[(size_t)(tm * 16 + r + 8 * kh) * Np + col] = v;
  }
}

// ---------------- edge message scatter: agg[dst] += lin[src] * norm ----------------
__global__ void k_agg_feat(const float* __restrict__ lin, const float* __restrict__ dis,
                           const int* __restrict__ src, const int* __restrict__ dst,
                           const int* __restrict__ em, float* __restrict__ agg, int E) {
  int wid  = blockIdx.x * (blockDim.x >> 5) + (threadIdx.x >> 5);
  int lane = threadIdx.x & 31;
  if (wid >= E) return;
  if (!em[wid]) return;
  int s = src[wid], d = dst[wid];
  float nrm = dis[s] * dis[d];
  const float4 hv = ((const float4*)(lin + (size_t)s * 128))[lane];
  float* out = agg + (size_t)d * 128 + lane * 4;
  atomicAdd(out + 0, hv.x * nrm);
  atomicAdd(out + 1, hv.y * nrm);
  atomicAdd(out + 2, hv.z * nrm);
  atomicAdd(out + 3, hv.w * nrm);
}

// feat = [relu](agg + lin * dis^2 + bias)
__global__ void k_finalize(const float* __restrict__ lin, const float* __restrict__ agg,
                           const float* __restrict__ dis, const float* __restrict__ bias,
                           float* __restrict__ feat, int n, int relu) {
  int t = blockIdx.x * blockDim.x + threadIdx.x;
  if (t >= n * 128) return;
  int i = t >> 7, f = t & 127;
  float di = dis[i];
  float v = agg[t] + lin[t] * di * di + bias[f];
  feat[t] = relu ? fmaxf(v, 0.f) : v;
}

// hs[i] = dot(X[i,0:128], Ws[0:128]) — one wave per node, shuffle reduce
__global__ void k_score_lin(const float* __restrict__ X, const float* __restrict__ Ws,
                            float* __restrict__ hs, int n) {
  int wid  = blockIdx.x * (blockDim.x >> 5) + (threadIdx.x >> 5);
  int lane = threadIdx.x & 31;
  if (wid >= n) return;
  const float* row = X + (size_t)wid * 128;
  float v = row[lane] * Ws[lane] + row[lane + 32] * Ws[lane + 32]
          + row[lane + 64] * Ws[lane + 64] + row[lane + 96] * Ws[lane + 96];
  for (int off = 16; off > 0; off >>= 1) v += __shfl_down(v, off);
  if (lane == 0) hs[wid] = v;
}

__global__ void k_agg_scalar(const float* __restrict__ hs, const float* __restrict__ dis,
                             const int* __restrict__ src, const int* __restrict__ dst,
                             const int* __restrict__ em, float* __restrict__ sagg, int E) {
  int e = blockIdx.x * blockDim.x + threadIdx.x;
  if (e < E && em[e]) {
    int s = src[e], d = dst[e];
    atomicAdd(&sagg[d], hs[s] * dis[s] * dis[d]);
  }
}

__global__ void k_score_final(const float* __restrict__ sagg, const float* __restrict__ hs,
                              const float* __restrict__ dis, const float* __restrict__ bs,
                              float* __restrict__ score, int n) {
  int i = blockIdx.x * blockDim.x + threadIdx.x;
  if (i < n) { float d = dis[i]; score[i] = sagg[i] + hs[i] * d * d + bs[0]; }
}

// per-graph top-k (selection, lowest-index tie-break like lax.top_k)
__global__ void k_topk(const float* __restrict__ score, int* __restrict__ perm,
                       int cur, int k) {
  __shared__ float sv[1024];
  __shared__ int   sel[1024];
  __shared__ float rv[256];
  __shared__ int   ri[256];
  int b = blockIdx.x, tid = threadIdx.x;
  for (int i = tid; i < cur; i += blockDim.x) { sv[i] = score[b * cur + i]; sel[i] = 0; }
  __syncthreads();
  for (int j = 0; j < k; ++j) {
    float bv = NEG_BIG; int bi = cur;
    for (int i = tid; i < cur; i += blockDim.x)
      if (!sel[i]) { float v = sv[i]; if (v > bv) { bv = v; bi = i; } }
    rv[tid] = bv; ri[tid] = bi;
    __syncthreads();
    for (int s = 128; s > 0; s >>= 1) {
      if (tid < s) {
        if (rv[tid + s] > rv[tid] ||
            (rv[tid + s] == rv[tid] && ri[tid + s] < ri[tid])) {
          rv[tid] = rv[tid + s]; ri[tid] = ri[tid + s];
        }
      }
      __syncthreads();
    }
    if (tid == 0) { sel[ri[0]] = 1; perm[b * k + j] = b * cur + ri[0]; }
    __syncthreads();
  }
}

// pool[t,f] = feat[perm[t],f] * tanh(score[perm[t]])
__global__ void k_gather(const float* __restrict__ feat, const float* __restrict__ score,
                         const int* __restrict__ perm, float* __restrict__ pool, int total) {
  int t = blockIdx.x * blockDim.x + threadIdx.x;
  if (t >= total) return;
  int nn = t >> 7, f = t & 127;
  int p = perm[nn];
  pool[t] = feat[(size_t)p * 128 + f] * tanhf(score[p]);
}

__global__ void k_setid(const int* __restrict__ perm, int* __restrict__ newid, int total) {
  int t = blockIdx.x * blockDim.x + threadIdx.x;
  if (t < total) newid[perm[t]] = t;
}

__global__ void k_remap(const int* __restrict__ newid, int* __restrict__ src,
                        int* __restrict__ dst, int* __restrict__ em, int E) {
  int e = blockIdx.x * blockDim.x + threadIdx.x;
  if (e >= E) return;
  int ns = newid[src[e]], nd = newid[dst[e]];
  int ok = em[e] && ns >= 0 && nd >= 0;
  em[e] = ok;
  src[e] = ok ? ns : 0;
  dst[e] = ok ? nd : 0;
}

// z[b, 0:128] += max over k nodes, z[b, 128:256] += mean
__global__ void k_readout(const float* __restrict__ pool, float* __restrict__ z, int k) {
  int b = blockIdx.x, f = threadIdx.x;  // blockDim = 128
  float mx = NEG_BIG, sm = 0.f;
  for (int t = 0; t < k; ++t) {
    float v = pool[((size_t)(b * k + t)) * 128 + f];
    mx = fmaxf(mx, v); sm += v;
  }
  z[b * 256 + f] += mx;
  z[b * 256 + 128 + f] += sm / (float)k;
}

__global__ void k_logsoftmax(const float* __restrict__ logits, float* __restrict__ out) {
  int r = threadIdx.x;  // 64 threads; logits padded to stride 16
  float m = NEG_BIG;
  for (int c = 0; c < 10; ++c) m = fmaxf(m, logits[r * 16 + c]);
  float s = 0.f;
  for (int c = 0; c < 10; ++c) s += expf(logits[r * 16 + c] - m);
  float ls = logf(s);
  for (int c = 0; c < 10; ++c) out[r * 10 + c] = logits[r * 16 + c] - m - ls;
}

// ---------------- host orchestration ----------------
static inline void cvtA(const float* in, _Float16* out, int M, int K, int Kp, hipStream_t s) {
  int total = M * Kp;
  k_cvt_f16<<<(total + 255) / 256, 256, 0, s>>>(in, out, K, Kp, total);
}
static inline void packW(const float* W, _Float16* out, int K, int N, int Kp, int Np,
                         hipStream_t s) {
  int total = (Np / 16) * (Kp / 32) * 512;
  k_pack_w<<<(total + 255) / 256, 256, 0, s>>>(W, out, K, N, Kp / 32, total);
}
static inline void gemm(const _Float16* A16, const _Float16* Wp, const float* bias,
                        float* C, int M, int Kp, int Np, int relu, hipStream_t s) {
  dim3 g(Np / 16, M / 16);
  k_gemm_wmma<<<g, 32, 0, s>>>(A16, Wp, bias, C, Kp, Np, relu);
}
static inline void fillf(float* p, float v, int n, hipStream_t s) {
  k_fill_f32<<<(n + 255) / 256, 256, 0, s>>>(p, v, n);
}

extern "C" void kernel_launch(void* const* d_in, const int* in_sizes, int n_in,
                              void* d_out, int out_size, void* d_ws, size_t ws_size,
                              hipStream_t stream) {
  const float* x  = (const float*)d_in[0];
  const int*   ei = (const int*)d_in[1];
  const float* Wf[3]  = {(const float*)d_in[2], (const float*)d_in[4], (const float*)d_in[6]};
  const float* bf[3]  = {(const float*)d_in[3], (const float*)d_in[5], (const float*)d_in[7]};
  const float* Wsc[3] = {(const float*)d_in[8], (const float*)d_in[10], (const float*)d_in[12]};
  const float* bsc[3] = {(const float*)d_in[9], (const float*)d_in[11], (const float*)d_in[13]};
  const float* Wl1 = (const float*)d_in[14]; const float* bl1 = (const float*)d_in[15];
  const float* Wl2 = (const float*)d_in[16]; const float* bl2 = (const float*)d_in[17];
  const float* Wl3 = (const float*)d_in[18]; const float* bl3 = (const float*)d_in[19];

  const int E  = in_sizes[1] / 2;   // 1048576
  const int N0 = 65536;             // B*N
  const int curA[3] = {1024, 205, 41};
  const int kA[3]   = {205, 41, 9};

  // bump allocator over d_ws (~137 MB)
  char* w = (char*)d_ws;
  auto alloc = [&](size_t bytes) {
    char* p = w; w += (bytes + 255) & ~(size_t)255; return (void*)p;
  };
  int*   srcW  = (int*)alloc((size_t)E * 4);
  int*   dstW  = (int*)alloc((size_t)E * 4);
  int*   emW   = (int*)alloc((size_t)E * 4);
  float* lin   = (float*)alloc((size_t)N0 * 128 * 4);
  float* agg   = (float*)alloc((size_t)N0 * 128 * 4);
  float* feat  = (float*)alloc((size_t)N0 * 128 * 4);
  _Float16* a16 = (_Float16*)alloc((size_t)N0 * 128 * 2);
  float* pool  = (float*)alloc((size_t)64 * 205 * 128 * 4);
  float* deg   = (float*)alloc((size_t)N0 * 4);
  float* dis   = (float*)alloc((size_t)N0 * 4);
  float* hs    = (float*)alloc((size_t)N0 * 4);
  float* sagg  = (float*)alloc((size_t)N0 * 4);
  float* score = (float*)alloc((size_t)N0 * 4);
  int*   newid = (int*)alloc((size_t)N0 * 4);
  int*   perm  = (int*)alloc((size_t)64 * 205 * 4);
  float* z     = (float*)alloc((size_t)64 * 256 * 4);
  float* t1    = (float*)alloc((size_t)64 * 128 * 4);
  float* t2    = (float*)alloc((size_t)64 * 64 * 4);
  float* lgt   = (float*)alloc((size_t)64 * 16 * 4);
  float* bl3p  = (float*)alloc(16 * 4);
  _Float16* Wfp[3];
  Wfp[0] = (_Float16*)alloc((size_t)8 * 1 * 512 * 2);   // W1: Kp=32,  Np=128
  Wfp[1] = (_Float16*)alloc((size_t)8 * 4 * 512 * 2);   // W2: Kp=128, Np=128
  Wfp[2] = (_Float16*)alloc((size_t)8 * 4 * 512 * 2);   // W3
  _Float16* Wl1p = (_Float16*)alloc((size_t)8 * 8 * 512 * 2);  // Kp=256, Np=128
  _Float16* Wl2p = (_Float16*)alloc((size_t)4 * 4 * 512 * 2);  // Kp=128, Np=64
  _Float16* Wl3p = (_Float16*)alloc((size_t)1 * 2 * 512 * 2);  // Kp=64,  Np=16

  const int TB = 256;
  const int egrid  = (E + TB - 1) / TB;
  const int ewgrid = E / (TB / 32);

  // weight packing (small, once per call for determinism)
  packW(Wf[0], Wfp[0], 10, 128, 32, 128, stream);
  packW(Wf[1], Wfp[1], 128, 128, 128, 128, stream);
  packW(Wf[2], Wfp[2], 128, 128, 128, 128, stream);
  packW(Wl1, Wl1p, 256, 128, 256, 128, stream);
  packW(Wl2, Wl2p, 128, 64, 128, 64, stream);
  packW(Wl3, Wl3p, 64, 10, 64, 16, stream);
  k_pad_bias<<<1, 16, 0, stream>>>(bl3, bl3p, 10, 16);

  k_edge_init<<<egrid, TB, 0, stream>>>(ei, srcW, dstW, emW, E);
  fillf(z, 0.f, 64 * 256, stream);

  const float* xin = x;
  int fin = 10, finp = 32;
  for (int st = 0; st < 3; ++st) {
    int cur = curA[st], n = 64 * cur, k = kA[st];
    // symmetric-norm degrees (self-loop +1)
    fillf(deg, 1.0f, n, stream);
    k_deg_accum<<<egrid, TB, 0, stream>>>(dstW, emW, deg, E);
    k_rsqrt<<<(n + TB - 1) / TB, TB, 0, stream>>>(deg, dis, n);
    // feature GCN: lin = xin @ W
    cvtA(xin, a16, n, fin, finp, stream);
    gemm(a16, Wfp[st], nullptr, lin, n, finp, 128, 0, stream);
    fillf(agg, 0.f, n * 128, stream);
    k_agg_feat<<<ewgrid, TB, 0, stream>>>(lin, dis, srcW, dstW, emW, agg, E);
    k_finalize<<<(n * 128 + TB - 1) / TB, TB, 0, stream>>>(lin, agg, dis, bf[st], feat, n, 1);
    // score GCN (single channel)
    k_score_lin<<<(n + 7) / 8, TB, 0, stream>>>(feat, Wsc[st], hs, n);
    fillf(sagg, 0.f, n, stream);
    k_agg_scalar<<<egrid, TB, 0, stream>>>(hs, dis, srcW, dstW, emW, sagg, E);
    k_score_final<<<(n + TB - 1) / TB, TB, 0, stream>>>(sagg, hs, dis, bsc[st], score, n);
    // top-k per graph + gather with tanh gate
    k_topk<<<64, 256, 0, stream>>>(score, perm, cur, k);
    int tot = 64 * k * 128;
    k_gather<<<(tot + TB - 1) / TB, TB, 0, stream>>>(feat, score, perm, pool, tot);
    // filter_adj (not needed after last stage)
    if (st < 2) {
      k_fill_i32<<<(n + TB - 1) / TB, TB, 0, stream>>>(newid, -1, n);
      k_setid<<<(64 * k + TB - 1) / TB, TB, 0, stream>>>(perm, newid, 64 * k);
      k_remap<<<egrid, TB, 0, stream>>>(newid, srcW, dstW, emW, E);
    }
    // readout accumulates into z
    k_readout<<<64, 128, 0, stream>>>(pool, z, k);
    xin = pool; fin = 128; finp = 128;
  }

  // MLP head (all WMMA)
  cvtA(z, a16, 64, 256, 256, stream);
  gemm(a16, Wl1p, bl1, t1, 64, 256, 128, 1, stream);
  cvtA(t1, a16, 64, 128, 128, stream);
  gemm(a16, Wl2p, bl2, t2, 64, 128, 64, 1, stream);
  cvtA(t2, a16, 64, 64, 64, stream);
  gemm(a16, Wl3p, bl3p, lgt, 64, 64, 16, 0, stream);
  k_logsoftmax<<<1, 64, 0, stream>>>(lgt, (float*)d_out);
}
